// PredAlignUpsample_57827439674194
// MI455X (gfx1250) — matrix operand order repaired and checked by reading, not compiled, not run on hardware
//
#include <hip/hip_runtime.h>

// CDNA5 / gfx1250 implementation of PredAlignUpsample (CARAFE-style).
// Stage A: guide = w_compress @ feat + b  via v_wmma_f32_16x16x32_f16 (f16 in, f32 acc)
// Stage B (fused): 3x3 conv (64->25) as 9 shifted WMMA GEMMs + softmax(25) kept in
//                  LDS, immediately consumed by the 25-tap CARAFE gather (no mask
//                  round-trip through HBM).

typedef __attribute__((ext_vector_type(16))) _Float16 v16h;
typedef __attribute__((ext_vector_type(8)))  _Float16 v8h;
typedef __attribute__((ext_vector_type(8)))  float    v8f;

#define B_   4
#define CP   80
#define HLO  64
#define WLO  64
#define CF   256
#define H_   128
#define W_   128
#define CC   64
#define HW   (H_ * W_)     // 16384
#define HWLO (HLO * WLO)   // 4096

__device__ __forceinline__ v16h cat16(v8h lo, v8h hi) {
  return __builtin_shufflevector(lo, hi, 0,1,2,3,4,5,6,7,8,9,10,11,12,13,14,15);
}

// ---------------------------------------------------------------------------
// Kernel A: guide[b][o][p] = sum_c Wc[o][c] * feat[b][c][p] + bc[o]   (f16 out)
// Block = 256 thr (8 waves). Block tile: M=64 x N=256 pixels, K loop 8x32.
// ---------------------------------------------------------------------------
#define LDA 264   // f16 stride (row = 528 B, 16B aligned, conflict-free frags)
#define LDB 40    // f16 stride (row = 80 B,  16B aligned, conflict-free frags)

__global__ __launch_bounds__(256) void compress_gemm_wmma(
    const float* __restrict__ feat, const float* __restrict__ Wc,
    const float* __restrict__ bc, _Float16* __restrict__ guide)
{
  __shared__ __align__(16) _Float16 smA[64 * LDA];
  __shared__ __align__(16) _Float16 smB[256 * LDB];

  const int t    = threadIdx.x;
  const int lane = t & 31;
  const int wv   = t >> 5;
  const int b    = blockIdx.x >> 6;          // 64 blocks per batch image
  const int n0   = (blockIdx.x & 63) * 256;  // pixel base within image

  // Stage A (w_compress -> f16 LDS), rows 0..63, cols 0..255
  #pragma unroll
  for (int i = 0; i < 64; ++i)
    smA[i * LDA + t] = (_Float16)Wc[i * 256 + t];

  const int mt  = wv & 3;          // M tile 0..3
  const int ntb = (wv >> 2) * 8;   // N tile base 0 or 8
  const int lm  = lane & 15;
  const int lh  = lane >> 4;       // 0|1 half-wave

  v8f acc[8] = {};

  const float* featB = feat + (size_t)b * CF * HW + n0;

  for (int ks = 0; ks < 8; ++ks) {
    // Prefetch next K slice: 8 threads per row cover its 8 x 128B cachelines
    if (ks < 7)
      __builtin_prefetch(
          featB + (size_t)((ks + 1) * 32 + (t & 31)) * HW + (t >> 5) * 32, 0, 1);
    __syncthreads();
    // Stage B tile: channels ks*32..+31, 256 pixels, transposed [pix][ch] f16
    #pragma unroll
    for (int i = 0; i < 32; ++i) {
      float v = featB[(size_t)(ks * 32 + i) * HW + t];
      smB[t * LDB + i] = (_Float16)v;
    }
    __syncthreads();

    // A fragment: lanes0-15 K 0-7/16-23, lanes16-31 K 8-15/24-31 (rel. slice)
    const _Float16* ap = &smA[(mt * 16 + lm) * LDA + ks * 32 + lh * 8];
    v16h a = cat16(*(const v8h*)ap, *(const v8h*)(ap + 16));

    // B fragment: lanes0-15 K 0-7/16-23, lanes16-31 K 8-15/24-31, in [pix][ch]
    auto loadB = [&](int j) -> v16h {
      const _Float16* bp = &smB[((ntb + j) * 16 + lm) * LDB + lh * 8];
      return cat16(*(const v8h*)bp, *(const v8h*)(bp + 16));
    };

    // Software pipeline: issue DS loads for tile j+1 before consuming tile j
    v16h bb = loadB(0);
    #pragma unroll
    for (int j = 0; j < 8; ++j) {
      v16h bn;
      if (j < 7) bn = loadB(j + 1);
      acc[j] = __builtin_amdgcn_wmma_f32_16x16x32_f16(
          false, a, false, bb, (short)0, acc[j], false, false);
      if (j < 7) bb = bn;
    }
  }

  // Epilogue: C layout lane%16 = col, VGPR r = row r (+8 for lanes>=16)
  #pragma unroll
  for (int j = 0; j < 8; ++j) {
    const int n = n0 + (ntb + j) * 16 + lm;
    #pragma unroll
    for (int r = 0; r < 8; ++r) {
      const int m = mt * 16 + r + lh * 8;
      float v = acc[j][r] + bc[m];
      guide[((size_t)b * CC + m) * HW + n] = (_Float16)v;
    }
  }
}

// ---------------------------------------------------------------------------
// Kernel B (fused): per (b, y) row of 128 px:
//   Phase 1: raw = conv3x3(guide, w_enc) + b_enc via WMMA (M=32 pad of 25,
//            K=64 in 2x32, 9 taps), softmax(25) -> normalized mask in LDS.
//   Phase 2: out[b,c,y,x] = sum_{i,j} predpad[b,c,y/2+i,x/2+j]*mask[i*5+j]
//            (reflect pad=2), pred staged in LDS per 16-channel chunk.
// Block = 256 thr (8 waves).
// ---------------------------------------------------------------------------
#define LDG 32   // f16 ch-stride of guide tile (row = 64 B, 16B aligned)

__global__ __launch_bounds__(256) void enc_softmax_carafe(
    const _Float16* __restrict__ guide, const float* __restrict__ We,
    const float* __restrict__ be, const float* __restrict__ pred,
    float* __restrict__ out)
{
  __shared__ __align__(16) _Float16 smW[9 * 32 * 64];   // [tap][e(32)][c(64)]
  __shared__ __align__(16) _Float16 smG[3 * 130 * LDG]; // [row*130+col][ch32]

  const int t    = threadIdx.x;
  const int lane = t & 31;
  const int wv   = t >> 5;
  const int b    = blockIdx.x >> 7;
  const int y    = blockIdx.x & 127;

  // Stage weights once: smW[(tap*32+e)*64+c] = w_enc[e][c][dy][dx] (0-pad e>=25)
  for (int idx = t; idx < 9 * 32 * 64; idx += 256) {
    const int c   = idx & 63;
    const int e   = (idx >> 6) & 31;
    const int tap = idx >> 11;
    _Float16 v = (_Float16)0.f;
    if (e < 25) v = (_Float16)We[(e * 64 + c) * 9 + tap];
    smW[idx] = v;
  }

  const int lm = lane & 15;
  const int lh = lane >> 4;
  const int n0 = wv * 16;     // wave's pixel-column base

  v8f acc0 = {}, acc1 = {};

  for (int ks = 0; ks < 2; ++ks) {
    __syncthreads();
    // Stage guide halo tile: rows y-1..y+1, cols -1..128, ch ks*32..+31 ('SAME' 0-pad)
    for (int idx = t; idx < 3 * 32 * 130; idx += 256) {
      const int x    = idx % 130;
      const int rest = idx / 130;
      const int c    = rest & 31;
      const int rc   = rest >> 5;
      const int gy   = y + rc - 1;
      const int gx   = x - 1;
      _Float16 v = (_Float16)0.f;
      if (gy >= 0 && gy < H_ && gx >= 0 && gx < W_)
        v = guide[((size_t)b * CC + ks * 32 + c) * HW + gy * W_ + gx];
      smG[(rc * 130 + x) * LDG + c] = v;
    }
    __syncthreads();

    auto loadBG = [&](int tap) -> v16h {
      const int dy = tap / 3, dx = tap % 3;
      const _Float16* bp = &smG[(dy * 130 + n0 + lm + dx) * LDG + lh * 8];
      return cat16(*(const v8h*)bp, *(const v8h*)(bp + 16));
    };

    v16h bb = loadBG(0);
    #pragma unroll
    for (int tap = 0; tap < 9; ++tap) {
      v16h bn;
      if (tap < 8) bn = loadBG(tap + 1);
      #pragma unroll
      for (int mtl = 0; mtl < 2; ++mtl) {
        const _Float16* ap =
            &smW[(tap * 32 + mtl * 16 + lm) * 64 + ks * 32 + lh * 8];
        v16h a = cat16(*(const v8h*)ap, *(const v8h*)(ap + 16));
        if (mtl == 0)
          acc0 = __builtin_amdgcn_wmma_f32_16x16x32_f16(
              false, a, false, bb, (short)0, acc0, false, false);
        else
          acc1 = __builtin_amdgcn_wmma_f32_16x16x32_f16(
              false, a, false, bb, (short)0, acc1, false, false);
      }
      if (tap < 8) bb = bn;
    }
  }

  __syncthreads();                 // smG no longer needed: reuse as exchange
  float* ex = (float*)smG;         // [128 px][33] f32 (16,896 B <= 24,960 B)

  // C layout: lane%16 = pixel col, VGPR r = row r (+8 for lanes>=16)
  const int px = n0 + lm;
  #pragma unroll
  for (int r = 0; r < 8; ++r) {
    const int e0 = r + lh * 8;          // tile0 rows 0..15
    ex[px * 33 + e0] = acc0[r] + be[e0];
    const int e1 = 16 + r + lh * 8;     // tile1 rows 16..31 (keep e<25)
    if (e1 < 25) ex[px * 33 + e1] = acc1[r] + be[e1];
  }
  __syncthreads();

  if (t < 128) {                   // softmax over the 25 taps, normalize in LDS
    float v[25];
    float mx = -1e30f;
    #pragma unroll
    for (int e = 0; e < 25; ++e) { v[e] = ex[t * 33 + e]; mx = fmaxf(mx, v[e]); }
    float s = 0.f;
    #pragma unroll
    for (int e = 0; e < 25; ++e) { v[e] = __expf(v[e] - mx); s += v[e]; }
    const float inv = 1.f / s;
    #pragma unroll
    for (int e = 0; e < 25; ++e) ex[t * 33 + e] = v[e] * inv;
  }
  __syncthreads();

  // ---- Phase 2: CARAFE gather (mask lives in ex, pred staged in smW region) --
  float* smP = (float*)smW;        // [16][5][69] f32 = 22,080 B <= 36,864 B

  const int x    = t & 127;
  const int half = t >> 7;         // 2 channel-halves per pixel
  const int iy   = y >> 1;
  const int ix   = x >> 1;

  float m[25];
  #pragma unroll
  for (int e = 0; e < 25; ++e) m[e] = ex[x * 33 + e];

  for (int cb = 0; cb < CP; cb += 16) {
    __syncthreads();
    for (int idx = t; idx < 16 * 5 * 68; idx += 256) {
      const int q    = idx % 68;       // padded col 0..67
      const int rest = idx / 68;
      const int r    = rest % 5;       // row tap
      const int c    = rest / 5;
      int sy = iy + r - 2; sy = sy < 0 ? -sy : (sy > 63 ? 126 - sy : sy);
      int sx = q - 2;      sx = sx < 0 ? -sx : (sx > 63 ? 126 - sx : sx);
      smP[(c * 5 + r) * 69 + q] =
          pred[((size_t)(b * CP + cb + c)) * HWLO + sy * WLO + sx];
    }
    __syncthreads();

    #pragma unroll
    for (int cc = 0; cc < 8; ++cc) {
      const int c = half * 8 + cc;
      float s = 0.f;
      #pragma unroll
      for (int i = 0; i < 5; ++i)
        #pragma unroll
        for (int j = 0; j < 5; ++j)
          s += smP[(c * 5 + i) * 69 + ix + j] * m[i * 5 + j];
      out[((size_t)(b * CP + cb + c)) * HW + y * W_ + x] = s;
    }
  }
}

// ---------------------------------------------------------------------------
extern "C" void kernel_launch(void* const* d_in, const int* in_sizes, int n_in,
                              void* d_out, int out_size, void* d_ws, size_t ws_size,
                              hipStream_t stream) {
  const float* pred = (const float*)d_in[0];
  const float* feat = (const float*)d_in[1];
  const float* Wc   = (const float*)d_in[2];
  const float* bc   = (const float*)d_in[3];
  const float* We   = (const float*)d_in[4];
  const float* be   = (const float*)d_in[5];
  float* out = (float*)d_out;

  _Float16* guide = (_Float16*)d_ws;   // 4*64*16384 f16 = 8 MB scratch

  compress_gemm_wmma<<<B_ * (HW / 256), 256, 0, stream>>>(feat, Wc, bc, guide);
  enc_softmax_carafe<<<B_ * H_, 256, 0, stream>>>(guide, We, be, pred, out);
}
